// GraphAttentionLayer_23691039605126
// MI455X (gfx1250) — compile-verified, matrix operand order checked by reading, not run on hardware
//
#include <hip/hip_runtime.h>
#include <hip/hip_bf16.h>

typedef __attribute__((ext_vector_type(16))) _Float16 v16h;
typedef __attribute__((ext_vector_type(8)))  _Float16 v8h;
typedef __attribute__((ext_vector_type(8)))  float    v8f;

#define BB 4
#define NN 2048
#define FF 128
#define HH 4
#define DD 32
#define NEG_INF_F (-9.0e15f)
#define SLOPE 0.2f

// ---------------------------------------------------------------------------
// Kernel A: Wh = X @ W per (b,h); emit WhT in f16 ([bh][d][n]) for WMMA B
// operand, plus e_src/e_dst row scores (wave32 shuffle reduction over d).
// ---------------------------------------------------------------------------
__global__ __launch_bounds__(256) void gat_prep(
    const float* __restrict__ X, const float* __restrict__ W,
    const float* __restrict__ av,
    _Float16* __restrict__ WhT, float* __restrict__ es, float* __restrict__ ed)
{
  __shared__ float sW[FF * DD];    // 16 KB
  __shared__ float sX[8 * FF];     // 4 KB
  const int b  = blockIdx.z;
  const int hh = blockIdx.y;
  const int n0 = blockIdx.x * 8;
  const int t  = threadIdx.x;

  // Stage W[hh] (128x32 f32) and 8 rows of X into LDS.
  const float4* Wv  = (const float4*)(W + (size_t)hh * FF * DD);
  float4*       sWv = (float4*)sW;
#pragma unroll
  for (int k = 0; k < 4; ++k) sWv[t + 256 * k] = Wv[t + 256 * k];
  ((float4*)sX)[t] = ((const float4*)(X + ((size_t)b * NN + n0) * FF))[t];
  __syncthreads();

  const int nl = t >> 5;   // local row 0..7 (== wave id)
  const int d  = t & 31;   // output dim 0..31 (== lane id)
  float acc = 0.f;
#pragma unroll 4
  for (int f = 0; f < FF; ++f) acc += sX[nl * FF + f] * sW[f * DD + d];

  const int bh = b * HH + hh;
  WhT[((size_t)(bh * DD + d)) * NN + n0 + nl] = (_Float16)acc;

  float ps = acc * av[hh * 2 * DD + d];        // a_src[h][d]
  float pd = acc * av[hh * 2 * DD + DD + d];   // a_dst[h][d]
#pragma unroll
  for (int off = 16; off; off >>= 1) {
    ps += __shfl_xor(ps, off, 32);
    pd += __shfl_xor(pd, off, 32);
  }
  if (d == 0) {
    es[(size_t)bh * NN + n0 + nl] = ps;
    ed[(size_t)bh * NN + n0 + nl] = pd;
  }
}

// ---------------------------------------------------------------------------
// Kernel B: per (b,h,i) row: online softmax stats (max m, inv-sum 1/s)
// streaming adj row + e_dst with float4/int4 loads; wave32 combine.
// ---------------------------------------------------------------------------
__device__ __forceinline__ void online_upd(float& m, float& s, float v) {
  float nm = fmaxf(m, v);
  s = s * __expf(m - nm) + __expf(v - nm);
  m = nm;
}

__global__ __launch_bounds__(256) void gat_rowstats(
    const int* __restrict__ adj, const float* __restrict__ es,
    const float* __restrict__ ed, float* __restrict__ mrow,
    float* __restrict__ sinv)
{
  const int gid  = blockIdx.x * 8 + (threadIdx.x >> 5);   // row in [0, B*H*N)
  const int lane = threadIdx.x & 31;
  const int bh = gid / NN;
  const int i  = gid % NN;
  const float esrc = es[gid];
  const int*   arow = adj + (size_t)i * NN;
  const float* edp  = ed + (size_t)bh * NN;

  float m = NEG_INF_F, s = 0.f;
  for (int j0 = lane * 4; j0 < NN; j0 += 128) {
    int4   a4 = *(const int4*)(arow + j0);
    float4 e4 = *(const float4*)(edp + j0);
#define RS_ONE(AC, EC)                                                  \
    { float _s = esrc + (EC);                                           \
      float _lr = _s > 0.f ? _s : SLOPE * _s;                           \
      float _v = (AC) > 0 ? _lr : NEG_INF_F;                            \
      online_upd(m, s, _v); }
    RS_ONE(a4.x, e4.x) RS_ONE(a4.y, e4.y) RS_ONE(a4.z, e4.z) RS_ONE(a4.w, e4.w)
#undef RS_ONE
  }
#pragma unroll
  for (int off = 16; off; off >>= 1) {
    float m2 = __shfl_xor(m, off, 32);
    float s2 = __shfl_xor(s, off, 32);
    float nm = fmaxf(m, m2);
    s = s * __expf(m - nm) + s2 * __expf(m2 - nm);
    m = nm;
  }
  if (lane == 0) {
    mrow[gid] = m;
    sinv[gid] = 1.0f / s;
  }
}

// ---------------------------------------------------------------------------
// Kernel C: flash-style P@Wh with v_wmma_f32_16x16x32_f16.
// One wave per 16-row i-tile of one (b,h); loop j in chunks of 32.
// A-frag lane map: lane<16 -> row=lane, K {0..7,16..23};
//                  lane>=16 -> row=lane-16, K {8..15,24..31}.
// B-frag lane map: same K pattern, column d = lane&15 (+16 for acc1).
// ---------------------------------------------------------------------------
__global__ __launch_bounds__(256) void gat_attn(
    const int* __restrict__ adj, const float* __restrict__ es,
    const float* __restrict__ mrow, const float* __restrict__ sinv,
    const float* __restrict__ ed, const _Float16* __restrict__ WhT,
    float* __restrict__ out)
{
  const int wave = threadIdx.x >> 5;
  const int tile = blockIdx.x * 8 + wave;        // [0, B*H*N/16)
  const int lane = threadIdx.x & 31;
  const int bh = tile / (NN / 16);
  const int i0 = (tile % (NN / 16)) * 16;
  const int b  = bh / HH;
  const int hh = bh % HH;
  const int halfhi = lane >> 4;                  // 0: lanes 0-15, 1: lanes 16-31
  const int col    = lane & 15;

  const int arow_i = i0 + col;                   // this lane's A-matrix row
  const float esrc = es[(size_t)bh * NN + arow_i];
  const float mr   = mrow[(size_t)bh * NN + arow_i];
  const int*      arow = adj + (size_t)arow_i * NN;
  const float*    edp  = ed + (size_t)bh * NN;
  const _Float16* bp0  = WhT + ((size_t)(bh * DD + col)) * NN;
  const _Float16* bp1  = bp0 + (size_t)16 * NN;

  v8f acc0 = {};
  v8f acc1 = {};

#define PVAL(AIJ, EDV)                                                  \
  ({ float _s = esrc + (EDV);                                           \
     float _lr = _s > 0.f ? _s : SLOPE * _s;                            \
     float _v = (AIJ) > 0 ? _lr : NEG_INF_F;                            \
     __expf(_v - mr); })

  for (int j0 = 0; j0 < NN; j0 += 32) {
    const int base0 = j0 + (halfhi << 3);
    const int base1 = base0 + 16;

    int4 aA = *(const int4*)(arow + base0);
    int4 aB = *(const int4*)(arow + base0 + 4);
    int4 aC = *(const int4*)(arow + base1);
    int4 aD = *(const int4*)(arow + base1 + 4);
    float4 eA = *(const float4*)(edp + base0);
    float4 eB = *(const float4*)(edp + base0 + 4);
    float4 eC = *(const float4*)(edp + base1);
    float4 eD = *(const float4*)(edp + base1 + 4);

    v16h afrag;
    afrag[0]  = (_Float16)PVAL(aA.x, eA.x);
    afrag[1]  = (_Float16)PVAL(aA.y, eA.y);
    afrag[2]  = (_Float16)PVAL(aA.z, eA.z);
    afrag[3]  = (_Float16)PVAL(aA.w, eA.w);
    afrag[4]  = (_Float16)PVAL(aB.x, eB.x);
    afrag[5]  = (_Float16)PVAL(aB.y, eB.y);
    afrag[6]  = (_Float16)PVAL(aB.z, eB.z);
    afrag[7]  = (_Float16)PVAL(aB.w, eB.w);
    afrag[8]  = (_Float16)PVAL(aC.x, eC.x);
    afrag[9]  = (_Float16)PVAL(aC.y, eC.y);
    afrag[10] = (_Float16)PVAL(aC.z, eC.z);
    afrag[11] = (_Float16)PVAL(aC.w, eC.w);
    afrag[12] = (_Float16)PVAL(aD.x, eD.x);
    afrag[13] = (_Float16)PVAL(aD.y, eD.y);
    afrag[14] = (_Float16)PVAL(aD.z, eD.z);
    afrag[15] = (_Float16)PVAL(aD.w, eD.w);

    v8h b0lo = *(const v8h*)(bp0 + base0);
    v8h b0hi = *(const v8h*)(bp0 + base1);
    v8h b1lo = *(const v8h*)(bp1 + base0);
    v8h b1hi = *(const v8h*)(bp1 + base1);
    v16h bf0 = __builtin_shufflevector(b0lo, b0hi,
        0, 1, 2, 3, 4, 5, 6, 7, 8, 9, 10, 11, 12, 13, 14, 15);
    v16h bf1 = __builtin_shufflevector(b1lo, b1hi,
        0, 1, 2, 3, 4, 5, 6, 7, 8, 9, 10, 11, 12, 13, 14, 15);

    acc0 = __builtin_amdgcn_wmma_f32_16x16x32_f16(
        false, afrag, false, bf0, (short)0, acc0, false, false);
    acc1 = __builtin_amdgcn_wmma_f32_16x16x32_f16(
        false, afrag, false, bf1, (short)0, acc1, false, false);
  }
#undef PVAL

  // C/D layout: VGPR r -> row r (lanes 0-15) / row r+8 (lanes 16-31), N=col.
#pragma unroll
  for (int r = 0; r < 8; ++r) {
    const int gi = i0 + r + (halfhi ? 8 : 0);
    const float sc = sinv[(size_t)bh * NN + gi];
    float* op = out + ((size_t)(b * NN + gi) * HH + hh) * DD;
    op[col]      = acc0[r] * sc;
    op[col + 16] = acc1[r] * sc;
  }
}

// ---------------------------------------------------------------------------
extern "C" void kernel_launch(void* const* d_in, const int* in_sizes, int n_in,
                              void* d_out, int out_size, void* d_ws, size_t ws_size,
                              hipStream_t stream)
{
  (void)in_sizes; (void)n_in; (void)out_size; (void)ws_size;
  const float* X   = (const float*)d_in[0];   // h   (B,N,F)
  const int*   adj = (const int*)d_in[1];     // adj (N,N)
  const float* W   = (const float*)d_in[2];   // W   (H,F,D)
  const float* av  = (const float*)d_in[3];   // a   (H,2D,1)
  float* out = (float*)d_out;

  char* ws = (char*)d_ws;
  _Float16* WhT = (_Float16*)ws;                               // 2 MB
  float* es = (float*)(ws + (size_t)BB * HH * DD * NN * sizeof(_Float16));
  float* ed = es + (size_t)BB * HH * NN;
  float* mr = ed + (size_t)BB * HH * NN;
  float* si = mr + (size_t)BB * HH * NN;

  dim3 gA(NN / 8, HH, BB);
  gat_prep<<<gA, 256, 0, stream>>>(X, W, av, WhT, es, ed);

  gat_rowstats<<<BB * HH * NN / 8, 256, 0, stream>>>(adj, es, ed, mr, si);

  gat_attn<<<BB * HH * (NN / 16) / 8, 256, 0, stream>>>(adj, es, mr, si, ed,
                                                        WhT, out);
}
